// SDC_UDA_64527588655259
// MI455X (gfx1250) — matrix operand order repaired and checked by reading, not compiled, not run on hardware
//
#include <hip/hip_runtime.h>
#include <hip/hip_bf16.h>
#include <math.h>

// ============================================================================
// SDC-UDA-style video denoiser forward pass for gfx1250 (MI455X, wave32).
// All GEMM-shaped work (convs via implicit GEMM, attention QK^T / P*V) runs on
// v_wmma_f32_16x16x32_f16.  f32 activations in HBM, f16 operands for the
// matrix units, f32 accumulation.
// Conv kernel is templated on (mode, KH, KW) so the im2col index decode uses
// compile-time constant div/mod (mul+shift) and fully unrolls.
// ============================================================================

typedef _Float16 half_t;
typedef __attribute__((ext_vector_type(16))) _Float16 v16h;
typedef __attribute__((ext_vector_type(8)))  _Float16 v8h;
typedef __attribute__((ext_vector_type(8)))  float    v8f;

#define WAVES_PER_BLOCK 8   // 256 threads = 8 wave32

// ---------------------------------------------------------------------------
// WMMA fragment helpers (A/B: contiguous k order, same in both operands; C/D per CDNA5 ISA 7.12.2)
//   A (16x32 f16):  M = lane%16 ;  K(h) = h + (lane/16)*16
//     -> two contiguous 8-half (16B) chunks per lane.
//   B (32x16 f16):  N = lane%16 ;  K(h) = h + (lane/16)*16
//     -> one contiguous 16-half (32B) chunk per lane.
//   C/D (16x16 f32): M = r + (lane/16)*8 ; N = lane%16
// ---------------------------------------------------------------------------
static __device__ __forceinline__ v16h load_a_frag(const half_t* __restrict__ row,
                                                   int k0, int hi) {
  v8h lo = *(const v8h*)(row + k0 + hi * 16);
  v8h up = *(const v8h*)(row + k0 + 8 + hi * 16);
  v16h a;
#pragma unroll
  for (int i = 0; i < 8; ++i) { a[i] = lo[i]; a[8 + i] = up[i]; }
  return a;
}

// ---------------------------------------------------------------------------
// Generic implicit-GEMM convolution with WMMA.
//   in : f32 NCHW (Nimg, Cin, Hin, Win)
//   w16: f16 packed [CoPad][Kpad], Ktot = Cin*KH*KW, zero padded
//   out: f32 NCHW (Nimg, Co, Ho, Wo)
// TMODE 0: normal conv (stride, pad).
// TMODE 1: ConvTranspose2d(k=3,s=2,p=1,output_padding=1) expressed as a
//          stride-1 conv over the 2x-dilated, (1,2)-padded input.
// KHC/KWC: compile-time kernel extents (index decode strength-reduced).
// ---------------------------------------------------------------------------
template <int TMODE, int KHC, int KWC>
__global__ void k_conv_wmma(const float* __restrict__ in,
                            const half_t* __restrict__ w16,
                            const float* __restrict__ bias,
                            float* __restrict__ out,
                            int Nimg, int Cin, int Hin, int Win,
                            int Co, int Ho, int Wo,
                            int stride, int pad,
                            int Kpad, int relu) {
  const int lane = threadIdx.x & 31;
  const int hi   = lane >> 4;
  const int lj   = lane & 15;
  const int Ntot = Nimg * Ho * Wo;
  const int tilesM = (Co + 15) >> 4;
  const int tilesN = (Ntot + 15) >> 4;
  int wid = blockIdx.x * WAVES_PER_BLOCK + (threadIdx.x >> 5);
  if (wid >= tilesM * tilesN) return;          // uniform per wave
  const int tm = wid % tilesM;
  const int tn = wid / tilesM;

  constexpr int KHW = KHC * KWC;
  const int Ktot = Cin * KHW;

  const half_t* arow = w16 + (size_t)(tm * 16 + lj) * Kpad;

  const int nb = tn * 16 + lj;
  const bool nvalid = nb < Ntot;
  int img = 0, oh = 0, ow = 0;
  if (nvalid) {
    img = nb / (Ho * Wo);
    int r = nb - img * (Ho * Wo);
    oh = r / Wo; ow = r - oh * Wo;
  }
  const float* inimg = in + (size_t)img * Cin * Hin * Win;

  v8f acc = {};
  for (int k0 = 0; k0 < Kpad; k0 += 32) {
    v16h a = load_a_frag(arow, k0, hi);
    v16h b;
    const int kb = k0 + hi * 16;
#pragma unroll
    for (int h = 0; h < 16; ++h) {
      float v = 0.0f;
      int k = kb + h;
      if (nvalid && k < Ktot) {
        int ci, kh, kw;
        if (KHW == 1) {
          ci = k; kh = 0; kw = 0;
        } else {
          ci = k / KHW;                 // constant divisor -> mul+shift
          int r = k - ci * KHW;
          kh = r / KWC;
          kw = r - kh * KWC;
        }
        int ih, iw; bool ok;
        if (TMODE == 0) {
          ih = oh * stride + kh - pad;
          iw = ow * stride + kw - pad;
          ok = (unsigned)ih < (unsigned)Hin && (unsigned)iw < (unsigned)Win;
        } else {
          int a2 = oh + kh - 1, b2 = ow + kw - 1;
          ok = (a2 >= 0) && !(a2 & 1) && (b2 >= 0) && !(b2 & 1);
          ih = a2 >> 1; iw = b2 >> 1;
          ok = ok && (ih < Hin) && (iw < Win);
        }
        if (ok) v = inimg[((size_t)ci * Hin + ih) * Win + iw];
      }
      b[h] = (half_t)v;
    }
    acc = __builtin_amdgcn_wmma_f32_16x16x32_f16(false, a, false, b,
                                                 (short)0, acc, false, false);
  }

#pragma unroll
  for (int r = 0; r < 8; ++r) {
    int co = tm * 16 + r + hi * 8;
    if (co < Co && nvalid) {
      float v = acc[r] + (bias ? bias[co] : 0.0f);
      if (relu) v = fmaxf(v, 0.0f);
      out[(((size_t)img * Co + co) * Ho + oh) * Wo + ow] = v;
    }
  }
}

// ---------------------------------------------------------------------------
// Batched NT GEMM with WMMA:  C[M][N] = A[M][Kd] * B[N][Kd]^T
// A,B f16 row-major (lda/ldb multiples of 32), C f32.  blockIdx.y = batch.
// Rows of A and B must be allocated padded to multiples of 16.
// ---------------------------------------------------------------------------
__global__ void k_gemm_nt_wmma(const half_t* __restrict__ A,
                               const half_t* __restrict__ Bm,
                               float* __restrict__ C,
                               int M, int N, int Kd,
                               int lda, int ldb, int ldc,
                               long long sA, long long sB, long long sC) {
  A  += (size_t)blockIdx.y * sA;
  Bm += (size_t)blockIdx.y * sB;
  C  += (size_t)blockIdx.y * sC;
  const int lane = threadIdx.x & 31;
  const int hi = lane >> 4, lj = lane & 15;
  const int tilesM = (M + 15) >> 4, tilesN = (N + 15) >> 4;
  int wid = blockIdx.x * WAVES_PER_BLOCK + (threadIdx.x >> 5);
  if (wid >= tilesM * tilesN) return;
  const int tm = wid % tilesM, tn = wid / tilesM;
  const half_t* arow = A  + (size_t)(tm * 16 + lj) * lda;
  const half_t* brow = Bm + (size_t)(tn * 16 + lj) * ldb;
  v8f acc = {};
  for (int k0 = 0; k0 < Kd; k0 += 32) {
    v16h a = load_a_frag(arow, k0, hi);
    v16h b = *(const v16h*)(brow + k0 + hi * 16);
    acc = __builtin_amdgcn_wmma_f32_16x16x32_f16(false, a, false, b,
                                                 (short)0, acc, false, false);
  }
#pragma unroll
  for (int r = 0; r < 8; ++r) {
    int m = tm * 16 + r + hi * 8, n = tn * 16 + lj;
    if (m < M && n < N) C[(size_t)m * ldc + n] = acc[r];
  }
}

// ---------------------------------------------------------------------------
// Weight packing to f16 [CoPad][Kpad]
// ---------------------------------------------------------------------------
__global__ void k_pack_w16(const float* __restrict__ w, half_t* __restrict__ o,
                           int Co, int Ktot, int Kpad, int tot) {
  int i = blockIdx.x * blockDim.x + threadIdx.x;
  if (i >= tot) return;
  int co = i / Kpad, k = i - co * Kpad;
  float v = (co < Co && k < Ktot) ? w[(size_t)co * Ktot + k] : 0.0f;
  o[i] = (half_t)v;
}

// ConvTranspose weights [Ci][Co][3][3] -> flipped, transposed [CoPad][Kpad]
__global__ void k_pack_wt16(const float* __restrict__ w, half_t* __restrict__ o,
                            int Ci, int Co, int Kpad, int tot) {
  int i = blockIdx.x * blockDim.x + threadIdx.x;
  if (i >= tot) return;
  int co = i / Kpad, k = i - co * Kpad;
  int Ktot = Ci * 9;
  float v = 0.0f;
  if (co < Co && k < Ktot) {
    int ci = k / 9, r = k - ci * 9, kh = r / 3, kw = r - kh * 3;
    v = w[(((size_t)ci * Co + co) * 3 + (2 - kh)) * 3 + (2 - kw)];
  }
  o[i] = (half_t)v;
}

// ---------------------------------------------------------------------------
// InstanceNorm2d (biased var, eps inside sqrt) + optional ReLU, in-place.
// One block per (n,c) plane.
// ---------------------------------------------------------------------------
__global__ void k_inorm(float* __restrict__ x, int HW, int relu) {
  __shared__ float s1[256], s2[256];
  __shared__ float mb, ib;
  float* plane = x + (size_t)blockIdx.x * HW;
  float s = 0.0f, q = 0.0f;
  for (int i = threadIdx.x; i < HW; i += blockDim.x) {
    float v = plane[i]; s += v; q += v * v;
  }
  s1[threadIdx.x] = s; s2[threadIdx.x] = q;
  __syncthreads();
  for (int st = blockDim.x >> 1; st > 0; st >>= 1) {
    if (threadIdx.x < st) {
      s1[threadIdx.x] += s1[threadIdx.x + st];
      s2[threadIdx.x] += s2[threadIdx.x + st];
    }
    __syncthreads();
  }
  if (threadIdx.x == 0) {
    float m = s1[0] / HW;
    float var = s2[0] / HW - m * m;
    mb = m; ib = rsqrtf(var + 1e-5f);
  }
  __syncthreads();
  float m = mb, inv = ib;
  for (int i = threadIdx.x; i < HW; i += blockDim.x) {
    float v = (plane[i] - m) * inv;
    if (relu) v = fmaxf(v, 0.0f);
    plane[i] = v;
  }
}

// custom channel LayerNorm: (x-m)/(std+eps)*g+b  (eps added to std)
__global__ void k_cnorm(const float* __restrict__ in, float* __restrict__ out,
                        const float* __restrict__ g, const float* __restrict__ b,
                        int N, int C, int HW) {
  int idx = blockIdx.x * blockDim.x + threadIdx.x;
  if (idx >= N * HW) return;
  int n = idx / HW, s = idx - n * HW;
  const float* p = in + (size_t)n * C * HW + s;
  float m = 0.0f;
  for (int c = 0; c < C; ++c) m += p[(size_t)c * HW];
  m /= C;
  float v = 0.0f;
  for (int c = 0; c < C; ++c) { float d = p[(size_t)c * HW] - m; v += d * d; }
  v /= C;
  float inv = 1.0f / (sqrtf(v) + 1e-5f);
  float* o = out + (size_t)n * C * HW + s;
  for (int c = 0; c < C; ++c)
    o[(size_t)c * HW] = (p[(size_t)c * HW] - m) * inv * g[c] + b[c];
}

__global__ void k_reflect_pad(const float* __restrict__ in, float* __restrict__ out,
                              int N, int C, int H, int W, int P) {
  int Ho = H + 2 * P, Wo = W + 2 * P;
  long long tot = (long long)N * C * Ho * Wo;
  long long i = (long long)blockIdx.x * blockDim.x + threadIdx.x;
  if (i >= tot) return;
  int x = (int)(i % Wo); long long r = i / Wo;
  int y = (int)(r % Ho); r /= Ho;
  int c = (int)(r % C);  int n = (int)(r / C);
  int iy = y - P; if (iy < 0) iy = -iy; if (iy >= H) iy = 2 * H - 2 - iy;
  int ix = x - P; if (ix < 0) ix = -ix; if (ix >= W) ix = 2 * W - 2 - ix;
  out[i] = in[(((size_t)n * C + c) * H + iy) * W + ix];
}

// nn.Unfold(3, stride=2, padding=1): (N,C,H,W) -> (N,C*9,Ho,Wo), c-major
__global__ void k_unfold3(const float* __restrict__ in, float* __restrict__ out,
                          int N, int C, int H, int W) {
  int Ho = (H - 1) / 2 + 1, Wo = (W - 1) / 2 + 1;
  long long tot = (long long)N * C * 9 * Ho * Wo;
  long long i = (long long)blockIdx.x * blockDim.x + threadIdx.x;
  if (i >= tot) return;
  int ow = (int)(i % Wo); long long r = i / Wo;
  int oh = (int)(r % Ho); r /= Ho;
  int col = (int)(r % 9); r /= 9;
  int c = (int)(r % C);   int n = (int)(r / C);
  int kh = col / 3, kw = col - kh * 3;
  int ih = oh * 2 + kh - 1, iw = ow * 2 + kw - 1;
  float v = 0.0f;
  if ((unsigned)ih < (unsigned)H && (unsigned)iw < (unsigned)W)
    v = in[(((size_t)n * C + c) * H + ih) * W + iw];
  out[i] = v;
}

// pack one image's heads: in [128][S] f32 -> out [8 heads][S][32] f16 (d>=16 zero)
__global__ void k_pack_heads(const float* __restrict__ in, half_t* __restrict__ o, int S) {
  int i = blockIdx.x * blockDim.x + threadIdx.x;
  int tot = 8 * S * 32;
  if (i >= tot) return;
  int h = i / (S * 32); int r = i - h * S * 32;
  int s = r / 32, d = r - s * 32;
  float v = (d < 16) ? in[(size_t)(h * 16 + d) * S + s] : 0.0f;
  o[i] = (half_t)v;
}

// pack V transposed: in [128][S] f32 -> out [8 heads][16][Sp] f16 (s>=S zero)
__global__ void k_pack_heads_T(const float* __restrict__ in, half_t* __restrict__ o,
                               int S, int Sp) {
  int i = blockIdx.x * blockDim.x + threadIdx.x;
  int tot = 8 * 16 * Sp;
  if (i >= tot) return;
  int h = i / (16 * Sp); int r = i - h * 16 * Sp;
  int d = r / Sp, s = r - d * Sp;
  float v = (s < S) ? in[(size_t)(h * 16 + d) * S + s] : 0.0f;
  o[i] = (half_t)v;
}

// f32 [rows][L] -> f16 [rows][Lp] zero-padded
__global__ void k_f16pad(const float* __restrict__ in, half_t* __restrict__ o,
                         int rows, int L, int Lp) {
  long long tot = (long long)rows * Lp;
  long long i = (long long)blockIdx.x * blockDim.x + threadIdx.x;
  if (i >= tot) return;
  int j = (int)(i % Lp); int r = (int)(i / Lp);
  o[i] = (half_t)((j < L) ? in[(size_t)r * L + j] : 0.0f);
}

// O [8 heads][S][16] f32 -> out [128][S] f32 (one image)
__global__ void k_unpack_heads(const float* __restrict__ O, float* __restrict__ out, int S) {
  int i = blockIdx.x * blockDim.x + threadIdx.x;
  int tot = 128 * S;
  if (i >= tot) return;
  int c = i / S, s = i - c * S;
  int h = c >> 4, d = c & 15;
  out[i] = O[((size_t)h * S + s) * 16 + d];
}

// row softmax with pre-scale; one wave per row (L <= 512)
__global__ void k_softmax(float* __restrict__ x, int rows, int L, float scale) {
  int row = blockIdx.x * WAVES_PER_BLOCK + (threadIdx.x >> 5);
  if (row >= rows) return;
  int lane = threadIdx.x & 31;
  float* p = x + (size_t)row * L;
  float tmp[16];
  int cnt = 0; float mx = -3.4e38f;
  for (int j = lane; j < L; j += 32) {
    float v = p[j] * scale; tmp[cnt++] = v; mx = fmaxf(mx, v);
  }
  for (int o = 16; o > 0; o >>= 1) mx = fmaxf(mx, __shfl_xor(mx, o, 32));
  float sum = 0.0f;
  for (int i = 0; i < cnt; ++i) { tmp[i] = __expf(tmp[i] - mx); sum += tmp[i]; }
  for (int o = 16; o > 0; o >>= 1) sum += __shfl_xor(sum, o, 32);
  float inv = 1.0f / sum;
  cnt = 0;
  for (int j = lane; j < L; j += 32) p[j] = tmp[cnt++] * inv;
}

// batch-slice attention: per (spatial s, head, query-frame b), attend over frames
__global__ void k_slice_attn(const float* __restrict__ qb, const float* __restrict__ kvb,
                             float* __restrict__ outb, int N, int HW, float scale) {
  int t = blockIdx.x * blockDim.x + threadIdx.x;
  int tot = HW * 8 * N;
  if (t >= tot) return;
  int s = t / (8 * N); int r = t - s * 8 * N;
  int h = r / N, bq = r - (r / N) * N;
  const float* q = qb + ((size_t)bq * 128 + h * 16) * HW + s;
  float sim[8]; float mx = -3.4e38f;
  for (int c = 0; c < 8; ++c) {
    const float* kk = kvb + ((size_t)c * 256 + h * 16) * HW + s;
    float d = 0.0f;
    for (int dd = 0; dd < 16; ++dd) d += q[(size_t)dd * HW] * kk[(size_t)dd * HW];
    sim[c] = d * scale; mx = fmaxf(mx, sim[c]);
  }
  float sum = 0.0f;
  for (int c = 0; c < 8; ++c) { sim[c] = __expf(sim[c] - mx); sum += sim[c]; }
  float inv = 1.0f / sum;
  for (int dd = 0; dd < 16; ++dd) {
    float o = 0.0f;
    for (int c = 0; c < 8; ++c)
      o += sim[c] * kvb[((size_t)c * 256 + 128 + h * 16 + dd) * HW + s];
    outb[((size_t)bq * 128 + h * 16 + dd) * HW + s] = o * inv;
  }
}

__global__ void k_add(float* __restrict__ dst, const float* __restrict__ src, long long n) {
  long long i = (long long)blockIdx.x * blockDim.x + threadIdx.x;
  if (i < n) dst[i] += src[i];
}

__global__ void k_gelu(float* __restrict__ x, long long n) {
  long long i = (long long)blockIdx.x * blockDim.x + threadIdx.x;
  if (i < n) { float v = x[i]; x[i] = 0.5f * v * (1.0f + erff(v * 0.70710678f)); }
}

__global__ void k_dwconv3(const float* __restrict__ in, const float* __restrict__ w,
                          const float* __restrict__ b, float* __restrict__ out,
                          int N, int C, int H, int W) {
  long long tot = (long long)N * C * H * W;
  long long i = (long long)blockIdx.x * blockDim.x + threadIdx.x;
  if (i >= tot) return;
  int x = (int)(i % W); long long r = i / W;
  int y = (int)(r % H); r /= H;
  int c = (int)(r % C); int n = (int)(r / C);
  const float* p = in + ((size_t)n * C + c) * H * W;
  const float* wc = w + (size_t)c * 9;
  float acc = b[c];
  for (int kh = 0; kh < 3; ++kh)
    for (int kw = 0; kw < 3; ++kw) {
      int iy = y + kh - 1, ix = x + kw - 1;
      if ((unsigned)iy < (unsigned)H && (unsigned)ix < (unsigned)W)
        acc += p[(size_t)iy * W + ix] * wc[kh * 3 + kw];
    }
  out[i] = acc;
}

__global__ void k_concat(const float* __restrict__ a, const float* __restrict__ b,
                         float* __restrict__ out, int N, int Ca, int Cb, int HW) {
  long long tot = (long long)N * (Ca + Cb) * HW;
  long long i = (long long)blockIdx.x * blockDim.x + threadIdx.x;
  if (i >= tot) return;
  int s = (int)(i % HW); long long r = i / HW;
  int c = (int)(r % (Ca + Cb)); int n = (int)(r / (Ca + Cb));
  out[i] = (c < Ca) ? a[((size_t)n * Ca + c) * HW + s]
                    : b[((size_t)n * Cb + (c - Ca)) * HW + s];
}

// final 1x1 conv 64->1 + tanh -> d_out
__global__ void k_last2_tanh(const float* __restrict__ in, const float* __restrict__ w,
                             const float* __restrict__ b, float* __restrict__ out,
                             int N, int C, int HW) {
  int i = blockIdx.x * blockDim.x + threadIdx.x;
  if (i >= N * HW) return;
  int n = i / HW, s = i - n * HW;
  const float* p = in + (size_t)n * C * HW + s;
  float acc = b[0];
  for (int c = 0; c < C; ++c) acc += p[(size_t)c * HW] * w[c];
  out[i] = tanhf(acc);
}

// ============================================================================
// Host orchestration
// ============================================================================
static inline size_t alignup(size_t x, size_t a) { return (x + a - 1) & ~(a - 1); }
static inline int aln(int x, int a) { return (x + a - 1) / a * a; }

extern "C" void kernel_launch(void* const* d_in, const int* in_sizes, int n_in,
                              void* d_out, int out_size, void* d_ws, size_t ws_size,
                              hipStream_t stream) {
  (void)in_sizes; (void)n_in; (void)out_size; (void)ws_size;
  const dim3 B256(256);
  auto eg = [](long long tot) { return dim3((unsigned)((tot + 255) / 256)); };

  // ---- unpack inputs (setup_inputs dict order) ----
  int ix = 0;
  auto nf = [&]() { return (const float*)d_in[ix++]; };
  const float* x = nf();
  const float *c1_w = nf(), *c1_b = nf(), *c2_w = nf(), *c2_b = nf();
  const float *c3_w = nf(), *c3_b = nf(), *d1_w = nf(), *d1_b = nf();
  const float *embed_w = nf(), *embed_b = nf();
  struct LW {
    const float *n1g, *n1b, *q_w, *kv_w, *out_w, *qb_w, *kvb_w;
    const float *n2g, *n2b, *ff1_w, *ff1_b, *dw_w, *dw_b, *pw_w, *pw_b, *ff2_w, *ff2_b;
  } L[4];
  for (int l = 0; l < 4; ++l) {
    L[l].n1g = nf(); L[l].n1b = nf();
    L[l].q_w = nf(); L[l].kv_w = nf(); L[l].out_w = nf();
    L[l].qb_w = nf(); L[l].kvb_w = nf();
    L[l].n2g = nf(); L[l].n2b = nf();
    L[l].ff1_w = nf(); L[l].ff1_b = nf();
    L[l].dw_w = nf(); L[l].dw_b = nf();
    L[l].pw_w = nf(); L[l].pw_b = nf();
    L[l].ff2_w = nf(); L[l].ff2_b = nf();
  }
  const float *up1_w = nf(), *up1_b = nf(), *up2_w = nf(), *up2_b = nf();
  const float *dec1a_w = nf(), *dec1a_b = nf(), *dec1b_w = nf(), *dec1b_b = nf();
  const float *last1_w = nf(), *last1_b = nf(), *last2_w = nf(), *last2_b = nf();

  // ---- workspace bump allocator ----
  char* ws = (char*)d_ws;
  size_t off = 0;
  auto allocB = [&](size_t bytes) -> void* {
    void* p = ws + off; off = alignup(off + bytes, 256); return p;
  };
  auto allocF = [&](size_t n) { return (float*)allocB(n * sizeof(float)); };
  auto allocH = [&](size_t n) { return (half_t*)allocB(n * sizeof(half_t)); };

  // ---- pack all conv weights to f16 [CoPad][Kpad] ----
  auto packw = [&](const float* w, int Co, int Ktot) -> half_t* {
    int CoPad = aln(Co, 16), Kpad = aln(Ktot, 32);
    half_t* p = allocH((size_t)CoPad * Kpad);
    int tot = CoPad * Kpad;
    k_pack_w16<<<eg(tot), B256, 0, stream>>>(w, p, Co, Ktot, Kpad, tot);
    return p;
  };
  auto packwt = [&](const float* w, int Ci, int Co) -> half_t* {
    int CoPad = aln(Co, 16), Kpad = aln(Ci * 9, 32);
    half_t* p = allocH((size_t)CoPad * Kpad);
    int tot = CoPad * Kpad;
    k_pack_wt16<<<eg(tot), B256, 0, stream>>>(w, p, Ci, Co, Kpad, tot);
    return p;
  };

  half_t* wc1 = packw(c1_w, 32, 49);
  half_t* wc2 = packw(c2_w, 32, 288);
  half_t* wc3 = packw(c3_w, 32, 288);
  half_t* wd1 = packw(d1_w, 64, 288);
  half_t* wembed = packw(embed_w, 128, 576);
  half_t *wq[4], *wkv[4], *wout[4], *wqb[4], *wkvb[4], *wff1[4], *wpw[4], *wff2[4];
  for (int l = 0; l < 4; ++l) {
    wq[l]   = packw(L[l].q_w, 128, 128);
    wkv[l]  = packw(L[l].kv_w, 256, 512);
    wout[l] = packw(L[l].out_w, 128, 128);
    wqb[l]  = packw(L[l].qb_w, 128, 128);
    wkvb[l] = packw(L[l].kvb_w, 256, 128);
    wff1[l] = packw(L[l].ff1_w, 512, 128);
    wpw[l]  = packw(L[l].pw_w, 512, 512);
    wff2[l] = packw(L[l].ff2_w, 128, 512);
  }
  half_t* wup1 = packwt(up1_w, 128, 64);
  half_t* wup2 = packwt(up2_w, 64, 32);
  half_t* wdec1a = packw(dec1a_w, 64, 576);
  half_t* wdec1b = packw(dec1b_w, 64, 576);
  half_t* wlast1 = packw(last1_w, 64, 3136);

  // ---- activation buffers ----
  const size_t BIG = (size_t)8 * 64 * 166 * 166;   // 14.1M floats
  float* B1 = allocF(BIG);
  float* B2 = allocF(BIG);
  float* B3 = allocF(BIG);
  float* SKIP = allocF((size_t)8 * 32 * 160 * 160);
  const int HW40 = 1600;
  float* Y   = allocF((size_t)8 * 128 * HW40);
  float* X   = allocF((size_t)8 * 128 * HW40);
  float* TQ  = allocF((size_t)8 * 128 * HW40);
  float* TQB = allocF((size_t)8 * 128 * HW40);
  float* TAI = allocF((size_t)8 * 128 * HW40);
  float* TAB = allocF((size_t)8 * 128 * HW40);
  float* TO  = allocF((size_t)8 * 128 * HW40);
  float* TF2 = allocF((size_t)8 * 128 * HW40);
  float* TKV  = allocF((size_t)8 * 256 * 400);
  float* TKVB = allocF((size_t)8 * 256 * HW40);
  float* TH1 = allocF((size_t)8 * 512 * HW40);
  float* TH2 = allocF((size_t)8 * 512 * HW40);
  // attention per-image scratch
  half_t* Qp  = allocH((size_t)8 * 1600 * 32);
  half_t* Kp  = allocH((size_t)8 * 400 * 32);
  half_t* VpT = allocH((size_t)8 * 16 * 416);
  float*  SIM = allocF((size_t)8 * 1600 * 400);
  half_t* P16 = allocH((size_t)8 * 1600 * 416);
  float*  Obuf = allocF((size_t)8 * 1600 * 16);

  // ---- launch helpers ----
  auto convL = [&](const float* in, const half_t* w, const float* bias, float* out,
                   int Nimg, int Cin, int Hin, int Win, int Co, int Ho, int Wo,
                   int KH, int KW, int st, int pad, int Kpad, int tmode) {
    int tiles = ((Co + 15) / 16) * ((Nimg * Ho * Wo + 15) / 16);
    dim3 g((tiles + WAVES_PER_BLOCK - 1) / WAVES_PER_BLOCK);
    if (tmode == 1) {
      k_conv_wmma<1, 3, 3><<<g, B256, 0, stream>>>(in, w, bias, out, Nimg, Cin, Hin, Win,
                                                   Co, Ho, Wo, st, pad, Kpad, 0);
    } else if (KH == 1 && KW == 1) {
      k_conv_wmma<0, 1, 1><<<g, B256, 0, stream>>>(in, w, bias, out, Nimg, Cin, Hin, Win,
                                                   Co, Ho, Wo, st, pad, Kpad, 0);
    } else if (KH == 3 && KW == 3) {
      k_conv_wmma<0, 3, 3><<<g, B256, 0, stream>>>(in, w, bias, out, Nimg, Cin, Hin, Win,
                                                   Co, Ho, Wo, st, pad, Kpad, 0);
    } else if (KH == 7 && KW == 7) {
      k_conv_wmma<0, 7, 7><<<g, B256, 0, stream>>>(in, w, bias, out, Nimg, Cin, Hin, Win,
                                                   Co, Ho, Wo, st, pad, Kpad, 0);
    } else { // 2x2 (kv conv)
      k_conv_wmma<0, 2, 2><<<g, B256, 0, stream>>>(in, w, bias, out, Nimg, Cin, Hin, Win,
                                                   Co, Ho, Wo, st, pad, Kpad, 0);
    }
  };
  auto gemm = [&](const half_t* A, const half_t* Bm, float* C, int M, int N, int Kd,
                  int lda, int ldb, int ldc, long long sA, long long sB, long long sC,
                  int batch) {
    int tiles = ((M + 15) / 16) * ((N + 15) / 16);
    dim3 g((tiles + WAVES_PER_BLOCK - 1) / WAVES_PER_BLOCK, batch);
    k_gemm_nt_wmma<<<g, B256, 0, stream>>>(A, Bm, C, M, N, Kd, lda, ldb, ldc, sA, sB, sC);
  };
  auto inorm = [&](float* p, int planes, int HW) {
    k_inorm<<<dim3(planes), B256, 0, stream>>>(p, HW, 1);
  };

  // ========================= encoder =========================
  k_reflect_pad<<<eg((long long)8 * 1 * 166 * 166), B256, 0, stream>>>(x, B3, 8, 1, 160, 160, 3);
  convL(B3, wc1, c1_b, B1, 8, 1, 166, 166, 32, 160, 160, 7, 7, 1, 0, 64, 0);
  inorm(B1, 8 * 32, 25600);
  convL(B1, wc2, c2_b, B2, 8, 32, 160, 160, 32, 160, 160, 3, 3, 1, 1, 288, 0);
  inorm(B2, 8 * 32, 25600);
  convL(B2, wc3, c3_b, SKIP, 8, 32, 160, 160, 32, 160, 160, 3, 3, 1, 1, 288, 0);
  inorm(SKIP, 8 * 32, 25600);
  convL(SKIP, wd1, d1_b, B1, 8, 32, 160, 160, 64, 80, 80, 3, 3, 2, 1, 288, 0);
  inorm(B1, 8 * 64, 6400);
  k_unfold3<<<eg((long long)8 * 64 * 9 * 1600), B256, 0, stream>>>(B1, B2, 8, 64, 80, 80);
  convL(B2, wembed, embed_b, Y, 8, 576, 40, 40, 128, 40, 40, 1, 1, 1, 0, 576, 0);

  // ========================= transformer =========================
  const float scale = 0.25f;  // dh=16 -> 16^-0.5
  for (int l = 0; l < 4; ++l) {
    // ---- attention block ----
    k_cnorm<<<eg(8 * HW40), B256, 0, stream>>>(Y, X, L[l].n1g, L[l].n1b, 8, 128, HW40);
    convL(X, wq[l], nullptr, TQ, 8, 128, 40, 40, 128, 40, 40, 1, 1, 1, 0, 128, 0);
    convL(X, wkv[l], nullptr, TKV, 8, 128, 40, 40, 256, 20, 20, 2, 2, 2, 0, 512, 0);
    for (int n = 0; n < 8; ++n) {
      const float* qn = TQ + (size_t)n * 128 * HW40;
      const float* kn = TKV + (size_t)n * 256 * 400;
      k_pack_heads<<<eg(8 * 1600 * 32), B256, 0, stream>>>(qn, Qp, 1600);
      k_pack_heads<<<eg(8 * 400 * 32), B256, 0, stream>>>(kn, Kp, 400);
      k_pack_heads_T<<<eg(8 * 16 * 416), B256, 0, stream>>>(kn + (size_t)128 * 400, VpT, 400, 416);
      gemm(Qp, Kp, SIM, 1600, 400, 32, 32, 32, 400,
           1600LL * 32, 400LL * 32, 1600LL * 400, 8);
      k_softmax<<<dim3((8 * 1600 + WAVES_PER_BLOCK - 1) / WAVES_PER_BLOCK), B256, 0, stream>>>(
          SIM, 8 * 1600, 400, scale);
      k_f16pad<<<eg((long long)8 * 1600 * 416), B256, 0, stream>>>(SIM, P16, 8 * 1600, 400, 416);
      gemm(P16, VpT, Obuf, 1600, 16, 416, 416, 416, 16,
           1600LL * 416, 16LL * 416, 1600LL * 16, 8);
      k_unpack_heads<<<eg(128 * 1600), B256, 0, stream>>>(Obuf, TAI + (size_t)n * 128 * HW40, 1600);
    }
    convL(X, wqb[l], nullptr, TQB, 8, 128, 40, 40, 128, 40, 40, 1, 1, 1, 0, 128, 0);
    convL(X, wkvb[l], nullptr, TKVB, 8, 128, 40, 40, 256, 40, 40, 1, 1, 1, 0, 128, 0);
    k_slice_attn<<<eg(1600 * 8 * 8), B256, 0, stream>>>(TQB, TKVB, TAB, 8, HW40, scale);
    k_add<<<eg((long long)8 * 128 * HW40), B256, 0, stream>>>(TAI, TAB, (long long)8 * 128 * HW40);
    convL(TAI, wout[l], nullptr, TO, 8, 128, 40, 40, 128, 40, 40, 1, 1, 1, 0, 128, 0);
    k_add<<<eg((long long)8 * 128 * HW40), B256, 0, stream>>>(Y, TO, (long long)8 * 128 * HW40);
    // ---- mix feed-forward ----
    k_cnorm<<<eg(8 * HW40), B256, 0, stream>>>(Y, X, L[l].n2g, L[l].n2b, 8, 128, HW40);
    convL(X, wff1[l], L[l].ff1_b, TH1, 8, 128, 40, 40, 512, 40, 40, 1, 1, 1, 0, 128, 0);
    k_dwconv3<<<eg((long long)8 * 512 * HW40), B256, 0, stream>>>(TH1, L[l].dw_w, L[l].dw_b, TH2,
                                                                  8, 512, 40, 40);
    convL(TH2, wpw[l], L[l].pw_b, TH1, 8, 512, 40, 40, 512, 40, 40, 1, 1, 1, 0, 512, 0);
    k_gelu<<<eg((long long)8 * 512 * HW40), B256, 0, stream>>>(TH1, (long long)8 * 512 * HW40);
    convL(TH1, wff2[l], L[l].ff2_b, TF2, 8, 512, 40, 40, 128, 40, 40, 1, 1, 1, 0, 512, 0);
    k_add<<<eg((long long)8 * 128 * HW40), B256, 0, stream>>>(Y, TF2, (long long)8 * 128 * HW40);
  }

  // ========================= decoder =========================
  convL(Y, wup1, up1_b, B1, 8, 128, 40, 40, 64, 80, 80, 3, 3, 0, 0, aln(128 * 9, 32), 1);
  inorm(B1, 8 * 64, 6400);
  convL(B1, wup2, up2_b, B2, 8, 64, 80, 80, 32, 160, 160, 3, 3, 0, 0, aln(64 * 9, 32), 1);
  inorm(B2, 8 * 32, 25600);
  k_concat<<<eg((long long)8 * 64 * 25600), B256, 0, stream>>>(B2, SKIP, B1, 8, 32, 32, 25600);
  convL(B1, wdec1a, dec1a_b, B2, 8, 64, 160, 160, 64, 160, 160, 3, 3, 1, 1, 576, 0);
  inorm(B2, 8 * 64, 25600);
  convL(B2, wdec1b, dec1b_b, B1, 8, 64, 160, 160, 64, 160, 160, 3, 3, 1, 1, 576, 0);
  inorm(B1, 8 * 64, 25600);
  k_reflect_pad<<<eg((long long)8 * 64 * 166 * 166), B256, 0, stream>>>(B1, B3, 8, 64, 160, 160, 3);
  convL(B3, wlast1, last1_b, B2, 8, 64, 166, 166, 64, 160, 160, 7, 7, 1, 0, 3136, 0);
  inorm(B2, 8 * 64, 25600);
  k_last2_tanh<<<eg(8 * 25600), B256, 0, stream>>>(B2, last2_w, last2_b, (float*)d_out,
                                                   8, 64, 25600);
}
